// SparseExpertRouter_88605175316806
// MI455X (gfx1250) — matrix operand order repaired and checked by reading, not compiled, not run on hardware
//
#include <hip/hip_runtime.h>
#include <hip/hip_bf16.h>

// ---------------- problem constants ----------------
#define NB 4
#define NS 2048
#define ND 1024
#define NE 8
#define NH 512
#define NK 2
#define NTOK (NB * NS)            // 8192 tokens
#define NENT (NTOK * NK)          // 16384 routed entries
#define CAPE (NENT + NE * 16)     // padded entry capacity: 16512
#define CAPT (CAPE / 16)          // row-tile capacity: 1032

typedef __attribute__((ext_vector_type(16))) __bf16 v16bf;
typedef __attribute__((ext_vector_type(8)))  __bf16 bf16x8;
typedef __attribute__((ext_vector_type(8)))  float  v8f;

union V16 { v16bf v; bf16x8 h[2]; };

__device__ __forceinline__ unsigned short f2bf(float f) {
  unsigned u = __float_as_uint(f);
  u += 0x7FFFu + ((u >> 16) & 1u);         // round-to-nearest-even
  return (unsigned short)(u >> 16);
}

__device__ __forceinline__ bf16x8 ld8(const unsigned short* p) {
  return *reinterpret_cast<const bf16x8*>(p);
}

__device__ __forceinline__ float gelu_exact(float t) {
  return 0.5f * t * (1.0f + erff(t * 0.70710678118654752440f));
}

// ---------------- fp32 -> bf16 elementwise ----------------
__global__ __launch_bounds__(256) void cvt_bf16_kernel(const float* __restrict__ in,
                                                       unsigned short* __restrict__ out,
                                                       long n) {
  long i = (long)blockIdx.x * blockDim.x + threadIdx.x;
  long stride = (long)gridDim.x * blockDim.x;
  for (; i < n; i += stride) out[i] = f2bf(in[i]);
}

// ---------------- fp32 [R][C] -> bf16 [C][R] (per z slice) ----------------
__global__ __launch_bounds__(256) void transpose_bf16_kernel(const float* __restrict__ in,
                                                             unsigned short* __restrict__ out,
                                                             int R, int C) {
  __shared__ float tile[32][33];
  long slice = (long)blockIdx.z * R * C;
  int cb = blockIdx.x * 32;  // column base in source
  int rb = blockIdx.y * 32;  // row base in source
  for (int i = threadIdx.y; i < 32; i += 8) {
    int r = rb + i, c = cb + threadIdx.x;
    tile[i][threadIdx.x] = (r < R && c < C) ? in[slice + (long)r * C + c] : 0.0f;
  }
  __syncthreads();
  for (int i = threadIdx.y; i < 32; i += 8) {
    int orow = cb + i;            // output row = source column
    int ocol = rb + threadIdx.x;  // output col = source row
    if (orow < C && ocol < R)
      out[slice + (long)orow * R + ocol] = f2bf(tile[threadIdx.x][i]);
  }
}

// ---------------- gating: sigmoid + top-2 (one wave per token) ----------------
__global__ __launch_bounds__(256) void gating_kernel(const float* __restrict__ x,
                                                     const float* __restrict__ gateW,
                                                     const float* __restrict__ route_scale,
                                                     int* __restrict__ topk_idx,
                                                     float* __restrict__ topk_w) {
  int lane = threadIdx.x & 31;
  int wave = threadIdx.x >> 5;
  int t = blockIdx.x * 8 + wave;
  if (t >= NTOK) return;
  const float* xr = x + (size_t)t * ND;
  float acc[NE];
#pragma unroll
  for (int e = 0; e < NE; ++e) acc[e] = 0.0f;
  for (int d = lane; d < ND; d += 32) {
    float xv = xr[d];
    const float4* g = reinterpret_cast<const float4*>(gateW + (size_t)d * NE);
    float4 g0 = g[0], g1 = g[1];
    acc[0] += xv * g0.x; acc[1] += xv * g0.y; acc[2] += xv * g0.z; acc[3] += xv * g0.w;
    acc[4] += xv * g1.x; acc[5] += xv * g1.y; acc[6] += xv * g1.z; acc[7] += xv * g1.w;
  }
#pragma unroll
  for (int off = 16; off > 0; off >>= 1) {
#pragma unroll
    for (int e = 0; e < NE; ++e) acc[e] += __shfl_down(acc[e], off);
  }
  if (lane == 0) {
    float rs = route_scale[0];
    float s[NE];
#pragma unroll
    for (int e = 0; e < NE; ++e) s[e] = rs / (1.0f + expf(-acc[e]));
    int i0 = 0;
#pragma unroll
    for (int e = 1; e < NE; ++e) if (s[e] > s[i0]) i0 = e;
    int i1 = -1;
#pragma unroll
    for (int e = 0; e < NE; ++e) {
      if (e == i0) continue;
      if (i1 < 0 || s[e] > s[i1]) i1 = e;
    }
    float sum = s[i0] + s[i1];
    topk_idx[t * 2 + 0] = i0;
    topk_idx[t * 2 + 1] = i1;
    topk_w[t * 2 + 0] = s[i0] / sum;
    topk_w[t * 2 + 1] = s[i1] / sum;
  }
}

// ---------------- routing build: stable scatter into per-expert segments ----
__global__ __launch_bounds__(256) void route_build_kernel(const int* __restrict__ topk_idx,
                                                          const float* __restrict__ topk_w,
                                                          int* __restrict__ perm_tok,
                                                          float* __restrict__ perm_w,
                                                          int* __restrict__ tile_expert) {
  __shared__ int lcnt[256 * NE];
  __shared__ int segstart[NE];
  int tid = threadIdx.x;
  // init padded buffers every launch (graph-replay safe)
  for (int i = tid; i < CAPE; i += 256) { perm_tok[i] = -1; perm_w[i] = 0.0f; }
  for (int i = tid; i < CAPT; i += 256) tile_expert[i] = -1;
  // local histogram over my contiguous chunk (stable order)
  const int CH = NENT / 256;  // 64
  int s0 = tid * CH;
  int c[NE];
#pragma unroll
  for (int e = 0; e < NE; ++e) c[e] = 0;
  for (int i = 0; i < CH; ++i) c[topk_idx[s0 + i]]++;
#pragma unroll
  for (int e = 0; e < NE; ++e) lcnt[tid * NE + e] = c[e];
  __syncthreads();
  if (tid == 0) {
    int cnt[NE];
#pragma unroll
    for (int e = 0; e < NE; ++e) cnt[e] = 0;
    for (int th = 0; th < 256; ++th)
#pragma unroll
      for (int e = 0; e < NE; ++e) cnt[e] += lcnt[th * NE + e];
    int run = 0;
    for (int e = 0; e < NE; ++e) {
      segstart[e] = run;
      int padded = (cnt[e] + 15) & ~15;
      for (int tt = run >> 4; tt < (run + padded) >> 4; ++tt) tile_expert[tt] = e;
      run += padded;
    }
  }
  __syncthreads();
  // exclusive prefix of earlier chunks -> stable base per expert
  int base[NE];
#pragma unroll
  for (int e = 0; e < NE; ++e) base[e] = segstart[e];
  for (int th = 0; th < tid; ++th)
#pragma unroll
    for (int e = 0; e < NE; ++e) base[e] += lcnt[th * NE + e];
  // stable scatter of my chunk
  for (int i = 0; i < CH; ++i) {
    int idx = s0 + i;
    int e = topk_idx[idx];
    int p = base[e]++;
    perm_tok[p] = idx >> 1;  // token id (K == 2)
    perm_w[p] = topk_w[idx];
  }
}

// Software-pipelined 16x64-per-wave bf16 WMMA mainloop. Branch-free steady
// state: while the 4 WMMAs of K-step kk issue, the B fragment for the next
// subtile and the A fragment for kk+32 are already in flight. The final
// K-step is peeled so no per-iteration conditionals reach the hot loop.
__device__ __forceinline__ void wmma_mainloop(const unsigned short* __restrict__ Aptr,
                                              const unsigned short* const* wp,
                                              int a0, int b0, int Kdim, v8f* acc) {
  V16 aCur, aNxt, bCur, bNxt;
  aCur.h[0] = ld8(Aptr + a0);
  aCur.h[1] = ld8(Aptr + 16 + a0);
  bCur.h[0] = ld8(wp[0] + b0);
  bCur.h[1] = ld8(wp[0] + b0 + 16);
  int kk = 0;
  for (; kk < Kdim - 32; kk += 32) {
    aNxt.h[0] = ld8(Aptr + kk + 32 + a0);
    aNxt.h[1] = ld8(Aptr + kk + 48 + a0);
#pragma unroll
    for (int s = 0; s < 4; ++s) {
      if (s < 3) {
        bNxt.h[0] = ld8(wp[s + 1] + kk + b0);
        bNxt.h[1] = ld8(wp[s + 1] + kk + b0 + 16);
      } else {
        bNxt.h[0] = ld8(wp[0] + kk + 32 + b0);
        bNxt.h[1] = ld8(wp[0] + kk + 32 + b0 + 16);
      }
      acc[s] = __builtin_amdgcn_wmma_f32_16x16x32_bf16(
          false, aCur.v, false, bCur.v, (short)0, acc[s], false, false);
      bCur = bNxt;
    }
    aCur = aNxt;
  }
  // peeled final K-step
#pragma unroll
  for (int s = 0; s < 4; ++s) {
    if (s < 3) {
      bNxt.h[0] = ld8(wp[s + 1] + kk + b0);
      bNxt.h[1] = ld8(wp[s + 1] + kk + b0 + 16);
    }
    acc[s] = __builtin_amdgcn_wmma_f32_16x16x32_bf16(
        false, aCur.v, false, bCur.v, (short)0, acc[s], false, false);
    if (s < 3) bCur = bNxt;
  }
}

// ---------------- GEMM1: H = gelu(A @ Wt^T + bias), bf16 WMMA ---------------
__global__ __launch_bounds__(128) void gemm_gelu_kernel(const unsigned short* __restrict__ A,
                                                        const unsigned short* __restrict__ Wt,
                                                        const float* __restrict__ bias,
                                                        unsigned short* __restrict__ Hout,
                                                        const int* __restrict__ rows,
                                                        const int* __restrict__ tile_expert,
                                                        int Kdim, int Ncols) {
  int tile = blockIdx.x;
  int e = 0;
  if (tile_expert) { e = tile_expert[tile]; if (e < 0) return; }
  int lane = threadIdx.x & 31;
  int wave = threadIdx.x >> 5;
  int colBase = blockIdx.y * 256 + wave * 64;
  int m16 = lane & 15;
  int mrow = tile * 16 + m16;
  int arow = rows ? rows[mrow] : mrow;
  // Padded rows (arow < 0) are clamped to row 0: their h output is garbage
  // but is never consumed (GEMM2 skips stores where perm_tok < 0). This keeps
  // the mainloop branch-free (no predicated zero-fill of A fragments).
  if (arow < 0) arow = 0;
  const unsigned short* Aptr = A + (size_t)arow * Kdim;
  const unsigned short* We = Wt + (size_t)e * Ncols * Kdim;
  const float* be = bias + (size_t)e * Ncols;
  const unsigned short* wp[4];
#pragma unroll
  for (int s = 0; s < 4; ++s)
    wp[s] = We + (size_t)(colBase + s * 16 + m16) * Kdim;

  int a0 = (lane < 16) ? 0 : 8;
  int b0 = (lane < 16) ? 0 : 8;
  v8f acc[4] = {v8f{}, v8f{}, v8f{}, v8f{}};

  wmma_mainloop(Aptr, wp, a0, b0, Kdim, acc);

  int mhi = (lane < 16) ? 0 : 8;
#pragma unroll
  for (int s = 0; s < 4; ++s) {
    int col = colBase + s * 16 + m16;
    float bv = be[col];
#pragma unroll
    for (int j = 0; j < 8; ++j) {
      int grow = tile * 16 + j + mhi;
      float v = acc[s][j] + bv;
      Hout[(size_t)grow * Ncols + col] = f2bf(gelu_exact(v));
    }
  }
}

// ---------------- GEMM2: out (+)= w_row * (A @ Wt^T + bias) -----------------
__global__ __launch_bounds__(128) void gemm_out_kernel(const unsigned short* __restrict__ A,
                                                       const unsigned short* __restrict__ Wt,
                                                       const float* __restrict__ bias,
                                                       float* __restrict__ Out,
                                                       const int* __restrict__ tokens,
                                                       const float* __restrict__ wts,
                                                       const int* __restrict__ tile_expert,
                                                       int Kdim, int Ncols) {
  int tile = blockIdx.x;
  int e = 0;
  if (tile_expert) { e = tile_expert[tile]; if (e < 0) return; }
  int lane = threadIdx.x & 31;
  int wave = threadIdx.x >> 5;
  int colBase = blockIdx.y * 256 + wave * 64;
  int m16 = lane & 15;
  const unsigned short* Aptr = A + (size_t)(tile * 16 + m16) * Kdim;
  const unsigned short* We = Wt + (size_t)e * Ncols * Kdim;
  const float* be = bias + (size_t)e * Ncols;
  const unsigned short* wp[4];
#pragma unroll
  for (int s = 0; s < 4; ++s)
    wp[s] = We + (size_t)(colBase + s * 16 + m16) * Kdim;

  int a0 = (lane < 16) ? 0 : 8;
  int b0 = (lane < 16) ? 0 : 8;
  v8f acc[4] = {v8f{}, v8f{}, v8f{}, v8f{}};

  wmma_mainloop(Aptr, wp, a0, b0, Kdim, acc);

  int mhi = (lane < 16) ? 0 : 8;
#pragma unroll
  for (int j = 0; j < 8; ++j) {
    int grow = tile * 16 + j + mhi;
    int tok = -1;
    float w = 1.0f;
    if (tokens) {
      tok = tokens[grow];
      if (tok < 0) continue;
      w = wts[grow];
    }
#pragma unroll
    for (int s = 0; s < 4; ++s) {
      int col = colBase + s * 16 + m16;
      float v = acc[s][j] + be[col];
      if (tokens) {
        atomicAdd(&Out[(size_t)tok * Ncols + col], v * w);
      } else {
        Out[(size_t)grow * Ncols + col] = v;
      }
    }
  }
}

// ---------------- host-side launcher ----------------
extern "C" void kernel_launch(void* const* d_in, const int* in_sizes, int n_in,
                              void* d_out, int out_size, void* d_ws, size_t ws_size,
                              hipStream_t stream) {
  const float* x      = (const float*)d_in[0];
  const float* gate_W = (const float*)d_in[1];
  const float* W1     = (const float*)d_in[2];
  const float* b1     = (const float*)d_in[3];
  const float* W2     = (const float*)d_in[4];
  const float* b2     = (const float*)d_in[5];
  const float* Ws1    = (const float*)d_in[6];
  const float* bs1    = (const float*)d_in[7];
  const float* Ws2    = (const float*)d_in[8];
  const float* bs2    = (const float*)d_in[9];
  const float* rs     = (const float*)d_in[10];
  float* out = (float*)d_out;

  char* ws = (char*)d_ws;
  size_t off = 0;
  auto alloc = [&](size_t bytes) {
    void* p = ws + off;
    off += (bytes + 255) & ~(size_t)255;
    return p;
  };
  unsigned short* xb    = (unsigned short*)alloc((size_t)NTOK * ND * 2);
  unsigned short* W1bt  = (unsigned short*)alloc((size_t)NE * NH * ND * 2);  // [E][H][D]
  unsigned short* W2bt  = (unsigned short*)alloc((size_t)NE * ND * NH * 2);  // [E][D][H]
  unsigned short* Ws1bt = (unsigned short*)alloc((size_t)NH * ND * 2);       // [H][D]
  unsigned short* Ws2bt = (unsigned short*)alloc((size_t)ND * NH * 2);       // [D][H]
  unsigned short* hs    = (unsigned short*)alloc((size_t)NTOK * NH * 2);     // shared h
  unsigned short* hbuf  = (unsigned short*)alloc((size_t)CAPE * NH * 2);     // expert h
  int*   topk_idx = (int*)alloc((size_t)NENT * 4);
  float* topk_w   = (float*)alloc((size_t)NENT * 4);
  int*   perm_tok = (int*)alloc((size_t)CAPE * 4);
  float* perm_w   = (float*)alloc((size_t)CAPE * 4);
  int*   tile_exp = (int*)alloc((size_t)CAPT * 4);

  // 1) precision conversion / transposition (bf16, [Ncols][K] layout)
  cvt_bf16_kernel<<<4096, 256, 0, stream>>>(x, xb, (long)NTOK * ND);
  transpose_bf16_kernel<<<dim3(NH / 32, ND / 32, NE), dim3(32, 8), 0, stream>>>(W1, W1bt, ND, NH);
  transpose_bf16_kernel<<<dim3(ND / 32, NH / 32, NE), dim3(32, 8), 0, stream>>>(W2, W2bt, NH, ND);
  transpose_bf16_kernel<<<dim3(NH / 32, ND / 32, 1),  dim3(32, 8), 0, stream>>>(Ws1, Ws1bt, ND, NH);
  transpose_bf16_kernel<<<dim3(ND / 32, NH / 32, 1),  dim3(32, 8), 0, stream>>>(Ws2, Ws2bt, NH, ND);

  // 2) gating + routing metadata
  gating_kernel<<<NTOK / 8, 256, 0, stream>>>(x, gate_W, rs, topk_idx, topk_w);
  route_build_kernel<<<1, 256, 0, stream>>>(topk_idx, topk_w, perm_tok, perm_w, tile_exp);

  // 3) shared expert: h = gelu(x Ws1 + bs1); out = h Ws2 + bs2  (initializes out)
  gemm_gelu_kernel<<<dim3(NTOK / 16, NH / 256), 128, 0, stream>>>(
      xb, Ws1bt, bs1, hs, nullptr, nullptr, ND, NH);
  gemm_out_kernel<<<dim3(NTOK / 16, ND / 256), 128, 0, stream>>>(
      hs, Ws2bt, bs2, out, nullptr, nullptr, nullptr, NH, ND);

  // 4) routed experts (grouped GEMMs over permuted token rows), atomically
  //    accumulated into out with normalized gate weights.
  gemm_gelu_kernel<<<dim3(CAPT, NH / 256), 128, 0, stream>>>(
      xb, W1bt, b1, hbuf, perm_tok, tile_exp, ND, NH);
  gemm_out_kernel<<<dim3(CAPT, ND / 256), 128, 0, stream>>>(
      hbuf, W2bt, b2, out, perm_tok, perm_w, tile_exp, NH, ND);
}
